// MultiHeadedAttention_11458972745761
// MI455X (gfx1250) — matrix operand order repaired and checked by reading, not compiled, not run on hardware
//
#include <hip/hip_runtime.h>
#include <hip/hip_bf16.h>

// ---------------------------------------------------------------------------
// Multi-head attention for MI455X (gfx1250), bf16 WMMA pipeline.
//   mask -> lane-ordered bitmask: one u32 per (lane, 32-key chunk, 32-row qtile)
//   q  = (query @ Wq + bq) * DH^-0.5          -> bf16 [B,H,LQ,DH]
//   kv = memory @ Wkv + bkv                    -> K bf16 [B,H,LK,DH], V^T bf16 [B,H,DH,LK]
//   flash attention (online softmax, 32-key chunks, 32 q-rows/wave) -> ctx bf16 [B,LQ,D]
//   out = ctx @ Wo + bo                        -> f32 [B,LQ,D]
// ---------------------------------------------------------------------------

typedef __bf16 bf16_t;
typedef bf16_t v16bf __attribute__((ext_vector_type(16)));
typedef bf16_t v8bf  __attribute__((ext_vector_type(8)));
typedef float  v8f   __attribute__((ext_vector_type(8)));
typedef float  v4f   __attribute__((ext_vector_type(4)));
typedef unsigned int u32;

#define B_   4
#define H_   16
#define D_   1024
#define DH_  64
#define LQ_  2048
#define LK_  2048
#define QT_  (LQ_ / 32)   // 64 q-tiles (32 rows each)
#define KC_  (LK_ / 32)   // 64 key chunks (32 keys each)
#define NEGV (-1.0e20f)

static __device__ __forceinline__ v16bf make16(v8bf lo, v8bf hi) {
  v16bf r;
#pragma unroll
  for (int i = 0; i < 8; ++i) { r[i] = lo[i]; r[i + 8] = hi[i]; }
  return r;
}

static __device__ __forceinline__ v16bf cvt16(v4f a0, v4f a1, v4f a2, v4f a3) {
  v16bf r;
#pragma unroll
  for (int i = 0; i < 4; ++i) {
    r[i]      = (bf16_t)a0[i];
    r[i + 4]  = (bf16_t)a1[i];
    r[i + 8]  = (bf16_t)a2[i];
    r[i + 12] = (bf16_t)a3[i];
  }
  return r;
}

// --------------------------- weight transpose+convert ----------------------
__global__ void mha_wt_convert(const float* __restrict__ W, bf16_t* __restrict__ Wt,
                               int K, int N) {
  size_t i = (size_t)blockIdx.x * blockDim.x + threadIdx.x;
  if (i >= (size_t)K * N) return;
  int n = (int)(i % N);
  int k = (int)(i / N);
  Wt[(size_t)n * K + k] = (bf16_t)W[(size_t)k * N + n];
}

// --------------------------- lane-order mask packing -----------------------
// bits[(((b*QT_+qt)*KC_)+kc)*32 + lane]: bit (t*16 + u*8 + v) is the mask of
// score element (row = qt*32 + u*16 + 8*(lane>>4) + v,
//                col = kc*32 + t*16 + (lane&15))
// i.e. exactly the element held in accumulator slot (u, t, vgpr v) of `lane`.
__global__ void mha_mask_pack_lane(const unsigned char* __restrict__ mask,
                                   u32* __restrict__ bits) {
  size_t i = (size_t)blockIdx.x * blockDim.x + threadIdx.x;
  if (i >= (size_t)B_ * QT_ * KC_ * 32) return;
  const int lane = (int)(i & 31);
  const int kc   = (int)((i >> 5) % KC_);
  const int qt   = (int)(((i >> 5) / KC_) % QT_);
  const int b    = (int)(((i >> 5) / KC_) / QT_);
  const int m = lane & 15, hi = lane >> 4;
  u32 w = 0;
#pragma unroll
  for (int u = 0; u < 2; ++u)
#pragma unroll
    for (int t = 0; t < 2; ++t)
#pragma unroll
      for (int v = 0; v < 8; ++v) {
        const int row = qt * 32 + u * 16 + 8 * hi + v;
        const int col = kc * 32 + t * 16 + m;
        if (mask[((size_t)b * LQ_ + row) * LK_ + col]) w |= 1u << (t * 16 + u * 8 + v);
      }
  bits[i] = w;
}

// ------------------------------ projection GEMM ----------------------------
// One wave computes a 32x64 output tile: 2 row blocks x 4 col accumulators.
// MODE 0: Q  -> bf16 [B,H,LQ,DH], val=(acc+bias)*scale
// MODE 1: KV -> K bf16 [B,H,LK,DH] (cols<1024), V^T bf16 [B,H,DH,LK] (cols>=1024)
// MODE 2: O  -> f32 row-major [Mtot, Ntot]
template <int MODE, bool ABF16>
__global__ __launch_bounds__(128) void mha_gemm_wmma(
    const void* __restrict__ Aptr, const bf16_t* __restrict__ Wt,
    const float* __restrict__ bias, void* __restrict__ Out0,
    void* __restrict__ Out1, int Mtot, int Ntot, int Ktot, float scale) {
  const int lane  = threadIdx.x & 31;
  const int wave  = blockIdx.x * (blockDim.x >> 5) + (threadIdx.x >> 5);
  const int ngrps = Ntot >> 6;
  const int row0  = (wave / ngrps) << 5;
  const int col0  = (wave % ngrps) << 6;
  const int m     = lane & 15;
  const int hi    = lane >> 4;

  v8f acc[2][4];
#pragma unroll
  for (int u = 0; u < 2; ++u)
#pragma unroll
    for (int g = 0; g < 4; ++g)
#pragma unroll
      for (int v = 0; v < 8; ++v) acc[u][g][v] = 0.0f;

  const int kiters = Ktot >> 5;
  for (int kk = 0; kk < kiters; ++kk) {
    const int kb = kk << 5;
    v16bf afrag[2];
#pragma unroll
    for (int u = 0; u < 2; ++u) {
      if (ABF16) {
        const bf16_t* ap =
            (const bf16_t*)Aptr + (size_t)(row0 + u * 16 + m) * Ktot + kb + (hi ? 8 : 0);
        afrag[u] = make16(*(const v8bf*)ap, *(const v8bf*)(ap + 16));
        if (kk + 1 < kiters) __builtin_prefetch(ap + 32, 0, 3);
      } else {
        const float* ap =
            (const float*)Aptr + (size_t)(row0 + u * 16 + m) * Ktot + kb + (hi ? 8 : 0);
        afrag[u] = cvt16(*(const v4f*)ap, *(const v4f*)(ap + 4),
                         *(const v4f*)(ap + 16), *(const v4f*)(ap + 20));
        if (kk + 1 < kiters) __builtin_prefetch(ap + 32, 0, 3);
      }
    }
#pragma unroll
    for (int g = 0; g < 4; ++g) {
      const bf16_t* wp = Wt + (size_t)(col0 + g * 16 + m) * Ktot + kb + hi * 8;
      v16bf bfrag = make16(*(const v8bf*)wp, *(const v8bf*)(wp + 16));
      if (kk + 1 < kiters) __builtin_prefetch(wp + 32, 0, 3);
#pragma unroll
      for (int u = 0; u < 2; ++u) {
        acc[u][g] = __builtin_amdgcn_wmma_f32_16x16x32_bf16(
            false, afrag[u], false, bfrag, (short)0, acc[u][g], false, false);
      }
    }
  }

#pragma unroll
  for (int u = 0; u < 2; ++u)
#pragma unroll
    for (int g = 0; g < 4; ++g) {
      const int col = col0 + g * 16 + m;
      const float bcol = bias[col];
#pragma unroll
      for (int v = 0; v < 8; ++v) {
        const int r = row0 + u * 16 + v + 8 * hi;
        const float val = (acc[u][g][v] + bcol) * scale;
        if (MODE == 0) {
          const int b = r / LQ_, lq = r % LQ_;
          const int hd = col >> 6, d = col & 63;
          ((bf16_t*)Out0)[((size_t)(b * H_ + hd) * LQ_ + lq) * DH_ + d] = (bf16_t)val;
        } else if (MODE == 1) {
          const int b = r / LK_, lk = r % LK_;
          if (col < D_) {
            const int hd = col >> 6, d = col & 63;
            ((bf16_t*)Out0)[((size_t)(b * H_ + hd) * LK_ + lk) * DH_ + d] = (bf16_t)val;
          } else {
            const int c2 = col - D_;
            const int hd = c2 >> 6, d = c2 & 63;
            ((bf16_t*)Out1)[((size_t)(b * H_ + hd) * DH_ + d) * LK_ + lk] = (bf16_t)val;
          }
        } else {
          ((float*)Out0)[(size_t)r * Ntot + col] = val;
        }
      }
    }
}

// ------------------------------ flash attention ----------------------------
// One wave per (b, head, 32-row q tile). Mask: one coalesced b32 load per
// chunk (lane-ordered bitmask), tested with compile-time-constant bit masks
// -- no LDS permutes, no variable shifts, no extra waits in the hot loop.
__global__ __launch_bounds__(128) void mha_attn_wmma(
    const bf16_t* __restrict__ Qb, const bf16_t* __restrict__ Kb,
    const bf16_t* __restrict__ Vt, const u32* __restrict__ maskbits,
    bf16_t* __restrict__ Ctx) {
  __shared__ bf16_t plds[4 * 2 * 512];  // 2KB per wave (two 16x32 P tiles)

  const int lane   = threadIdx.x & 31;
  const int widx   = threadIdx.x >> 5;
  const int wave   = blockIdx.x * 4 + widx;
  const int bh     = wave / QT_;
  const int qt     = wave % QT_;
  const int b      = bh / H_;
  const int hd     = bh % H_;
  const int m      = lane & 15;
  const int hi     = lane >> 4;

  // Q A-fragments for both row blocks (d 0..31, 32..63), loaded once.
  v16bf aq[2][2];
#pragma unroll
  for (int u = 0; u < 2; ++u) {
    const bf16_t* qbase = Qb + ((size_t)bh * LQ_ + qt * 32 + u * 16 + m) * DH_;
#pragma unroll
    for (int f = 0; f < 2; ++f) {
      const bf16_t* qp = qbase + f * 32 + hi * 8;
      aq[u][f] = make16(*(const v8bf*)qp, *(const v8bf*)(qp + 16));
    }
  }

  v8f acc[2][4];
#pragma unroll
  for (int u = 0; u < 2; ++u)
#pragma unroll
    for (int g = 0; g < 4; ++g)
#pragma unroll
      for (int v = 0; v < 8; ++v) acc[u][g][v] = 0.0f;

  float mr[2][8], lr[2][8];
#pragma unroll
  for (int u = 0; u < 2; ++u)
#pragma unroll
    for (int v = 0; v < 8; ++v) { mr[u][v] = -3.0e38f; lr[u][v] = 0.0f; }

  bf16_t* myp = plds + widx * 1024;
  const bf16_t* kbase = Kb + (size_t)bh * LK_ * DH_;
  const bf16_t* vbase = Vt + (size_t)bh * DH_ * LK_;
  // lane-ordered mask words: stride 32 words per chunk, coalesced per wave
  const u32* mbase = maskbits + (((size_t)b * QT_ + qt) * KC_) * 32 + lane;

#pragma unroll 1
  for (int kc = 0; kc < KC_; ++kc) {
    const int kb0 = kc * 32;
    const u32 w = mbase[kc * 32];  // all 32 mask bits this lane needs

    // ---- K fragments for this chunk (shared by both row blocks) ----
    v16bf bk[2][2];
#pragma unroll
    for (int t = 0; t < 2; ++t)
#pragma unroll
      for (int f = 0; f < 2; ++f) {
        const bf16_t* kp = kbase + (size_t)(kb0 + t * 16 + m) * DH_ + f * 32 + hi * 8;
        bk[t][f] = make16(*(const v8bf*)kp, *(const v8bf*)(kp + 16));
      }
    if (kc + 1 < KC_) {
      __builtin_prefetch(kbase + (size_t)(kb0 + 32 + m) * DH_, 0, 3);
      __builtin_prefetch(vbase + (size_t)m * LK_ + kb0 + 32, 0, 3);
    }

    v16bf ap[2];
#pragma unroll
    for (int u = 0; u < 2; ++u) {
      // ---- S = Q K^T for 2 key sub-tiles ----
      v8f s[2];
#pragma unroll
      for (int t = 0; t < 2; ++t) {
#pragma unroll
        for (int v = 0; v < 8; ++v) s[t][v] = 0.0f;
#pragma unroll
        for (int f = 0; f < 2; ++f) {
          s[t] = __builtin_amdgcn_wmma_f32_16x16x32_bf16(
              false, aq[u][f], false, bk[t][f], (short)0, s[t], false, false);
        }
      }

      // ---- mask: constant-bit tests on the lane-ordered word ----
#pragma unroll
      for (int v = 0; v < 8; ++v) {
        if (w & (1u << (u * 8 + v)))        s[0][v] = NEGV;
        if (w & (1u << (16 + u * 8 + v)))   s[1][v] = NEGV;
      }

      // ---- row max (reduce across 16 lanes of each lane-half) ----
      float tmax[8];
#pragma unroll
      for (int v = 0; v < 8; ++v) tmax[v] = fmaxf(s[0][v], s[1][v]);
#pragma unroll
      for (int sh = 1; sh < 16; sh <<= 1)
#pragma unroll
        for (int v = 0; v < 8; ++v) tmax[v] = fmaxf(tmax[v], __shfl_xor(tmax[v], sh, 32));

      // ---- online rescale ----
      float esc[8];
#pragma unroll
      for (int v = 0; v < 8; ++v) {
        const float mn = fmaxf(mr[u][v], tmax[v]);
        esc[v] = __expf(mr[u][v] - mn);
        mr[u][v] = mn;
        lr[u][v] *= esc[v];
      }
#pragma unroll
      for (int g = 0; g < 4; ++g)
#pragma unroll
        for (int v = 0; v < 8; ++v) acc[u][g][v] *= esc[v];

      // ---- P = exp(S - m), row sums ----
      float rs[8];
#pragma unroll
      for (int v = 0; v < 8; ++v) {
        s[0][v] = __expf(s[0][v] - mr[u][v]);
        s[1][v] = __expf(s[1][v] - mr[u][v]);
        rs[v] = s[0][v] + s[1][v];
      }
#pragma unroll
      for (int sh = 1; sh < 16; sh <<= 1)
#pragma unroll
        for (int v = 0; v < 8; ++v) rs[v] += __shfl_xor(rs[v], sh, 32);
#pragma unroll
      for (int v = 0; v < 8; ++v) lr[u][v] += rs[v];

      // ---- stage P (C-layout) to LDS, reload in A-layout ----
#pragma unroll
      for (int t = 0; t < 2; ++t)
#pragma unroll
        for (int v = 0; v < 8; ++v) {
          const int r = v + 8 * hi;
          myp[u * 512 + r * 32 + t * 16 + m] = (bf16_t)s[t][v];
        }
      const bf16_t* pp = myp + u * 512 + m * 32 + hi * 8;
      ap[u] = make16(*(const v8bf*)pp, *(const v8bf*)(pp + 16));
    }

    // ---- V fragments (shared), ctx += P V for both row blocks ----
#pragma unroll
    for (int g = 0; g < 4; ++g) {
      const bf16_t* vp = vbase + (size_t)(g * 16 + m) * LK_ + kb0 + hi * 8;
      v16bf bv = make16(*(const v8bf*)vp, *(const v8bf*)(vp + 16));
#pragma unroll
      for (int u = 0; u < 2; ++u) {
        acc[u][g] = __builtin_amdgcn_wmma_f32_16x16x32_bf16(
            false, ap[u], false, bv, (short)0, acc[u][g], false, false);
      }
    }
  }

  // ---- normalize, write ctx bf16 merged-head [B, LQ, D] ----
#pragma unroll
  for (int u = 0; u < 2; ++u) {
    float inv[8];
#pragma unroll
    for (int v = 0; v < 8; ++v) inv[v] = 1.0f / lr[u][v];
#pragma unroll
    for (int g = 0; g < 4; ++g)
#pragma unroll
      for (int v = 0; v < 8; ++v) {
        const int r = u * 16 + v + 8 * hi;
        Ctx[((size_t)b * LQ_ + qt * 32 + r) * D_ + hd * DH_ + g * 16 + m] =
            (bf16_t)(acc[u][g][v] * inv[v]);
      }
  }
}

// ------------------------------- launcher ----------------------------------
extern "C" void kernel_launch(void* const* d_in, const int* in_sizes, int n_in,
                              void* d_out, int out_size, void* d_ws, size_t ws_size,
                              hipStream_t stream) {
  const float* query = (const float*)d_in[0];
  const float* memry = (const float*)d_in[1];
  const unsigned char* mask = (const unsigned char*)d_in[2];
  const float* Wq  = (const float*)d_in[3];
  const float* bq  = (const float*)d_in[4];
  const float* Wkv = (const float*)d_in[5];
  const float* bkv = (const float*)d_in[6];
  const float* Wo  = (const float*)d_in[7];
  const float* bo  = (const float*)d_in[8];
  float* out = (float*)d_out;

  const size_t nQ = (size_t)B_ * H_ * LQ_ * DH_;  // 8.39M elements each
  bf16_t* qb   = (bf16_t*)d_ws;
  bf16_t* kb   = qb + nQ;
  bf16_t* vt   = kb + nQ;
  bf16_t* ctx  = vt + nQ;
  bf16_t* WqT  = ctx + nQ;
  bf16_t* WkvT = WqT + (size_t)D_ * D_;
  bf16_t* WoT  = WkvT + (size_t)D_ * 2 * D_;
  u32*    mbits = (u32*)(WoT + (size_t)D_ * D_);  // 2MB

  const int thr = 256;
  // 1) weights -> transposed bf16; mask -> lane-ordered bitmask
  mha_wt_convert<<<(D_ * D_ + thr - 1) / thr, thr, 0, stream>>>(Wq, WqT, D_, D_);
  mha_wt_convert<<<(D_ * 2 * D_ + thr - 1) / thr, thr, 0, stream>>>(Wkv, WkvT, D_, 2 * D_);
  mha_wt_convert<<<(D_ * D_ + thr - 1) / thr, thr, 0, stream>>>(Wo, WoT, D_, D_);
  const int nwords = B_ * QT_ * KC_ * 32;
  mha_mask_pack_lane<<<(nwords + thr - 1) / thr, thr, 0, stream>>>(mask, mbits);

  // 2) Q projection (scale DH^-0.5 folded in)
  const int wavesQ = (B_ * LQ_ / 32) * (D_ / 64);        // 4096
  mha_gemm_wmma<0, false><<<wavesQ / 4, 128, 0, stream>>>(
      query, WqT, bq, qb, nullptr, B_ * LQ_, D_, D_, 0.125f);

  // 3) KV projection (K and transposed V)
  const int wavesKV = (B_ * LK_ / 32) * (2 * D_ / 64);   // 8192
  mha_gemm_wmma<1, false><<<wavesKV / 4, 128, 0, stream>>>(
      memry, WkvT, bkv, kb, vt, B_ * LK_, 2 * D_, D_, 1.0f);

  // 4) flash attention (32 q rows per wave)
  const int wavesA = B_ * H_ * QT_;                      // 4096
  mha_attn_wmma<<<wavesA / 4, 128, 0, stream>>>(qb, kb, vt, mbits, ctx);

  // 5) output projection -> f32
  mha_gemm_wmma<2, true><<<wavesQ / 4, 128, 0, stream>>>(
      ctx, WoT, bo, out, nullptr, B_ * LQ_, D_, D_, 1.0f);
}
